// EmotionAttention_69320772158288
// MI455X (gfx1250) — matrix operand order repaired and checked by reading, not compiled, not run on hardware
//
#include <hip/hip_runtime.h>
#include <hip/hip_bf16.h>
#include <math.h>

// Problem constants (from reference)
#define B_    32
#define T_    4096
#define QS_   512
#define KS_   512
#define BN_   256
#define DV_   512
#define TOPK_ 20

// GEMM tiling
#define MT  64          // T-rows per workgroup
#define KT  32          // K per WMMA step (bf16 16x16x32)
#define NSTEP (KS_/KT)  // 16 K-steps
#define LDA 48          // LDS row stride in bf16 elems (96 B, 16B-aligned rows, padded)

typedef __attribute__((ext_vector_type(16))) __bf16 bf16x16;
typedef __attribute__((ext_vector_type(8)))  __bf16 bf16x8;
typedef __attribute__((ext_vector_type(8)))  float  f32x8;

union BF16Frag { bf16x16 v; bf16x8 h[2]; };

#define LOG2E_F 1.4426950408889634f

// Branch-free tanh: ~5 VALU ops (abs, mul, v_exp_f32, add/sub, v_rcp_f32, mul, copysign)
__device__ __forceinline__ float fast_tanh(float x) {
    const float a = __builtin_fabsf(x);
    const float z = __builtin_amdgcn_exp2f(-2.0f * LOG2E_F * a);   // exp(-2|x|)
    const float t = (1.0f - z) * __builtin_amdgcn_rcpf(1.0f + z);
    return __builtin_copysignf(t, x);
}

__device__ __forceinline__ float fast_exp(float x) {               // exp(x), x <= 0 here
    return __builtin_amdgcn_exp2f(x * LOG2E_F);
}

__device__ __forceinline__ void cvt_store8(__bf16* dst, float4 v0, float4 v1) {
    dst[0] = (__bf16)v0.x; dst[1] = (__bf16)v0.y;
    dst[2] = (__bf16)v0.z; dst[3] = (__bf16)v0.w;
    dst[4] = (__bf16)v1.x; dst[5] = (__bf16)v1.y;
    dst[6] = (__bf16)v1.z; dst[7] = (__bf16)v1.w;
}

// ---------------------------------------------------------------------------
// Kernel 1: Whb[b][n] = query[b,:] . W[n,:] + bias[n]   (tiny, fp32 VALU)
// ---------------------------------------------------------------------------
__global__ __launch_bounds__(256) void whb_kernel(
    const float* __restrict__ query, const float* __restrict__ W,
    const float* __restrict__ bias, float* __restrict__ Whb)
{
    const int b = blockIdx.x;
    const int n = threadIdx.x;          // 256 = BN
    const float* q  = query + (size_t)b * QS_;
    const float* wr = W     + (size_t)n * QS_;
    float s = 0.f;
    #pragma unroll 4
    for (int k = 0; k < QS_; k += 4) {
        s += q[k]   * wr[k]   + q[k+1] * wr[k+1]
           + q[k+2] * wr[k+2] + q[k+3] * wr[k+3];
    }
    Whb[(size_t)b * BN_ + n] = s + bias[n];
}

// ---------------------------------------------------------------------------
// Kernel 2: Uv = keys @ U^T via bf16 WMMA, fused energy epilogue.
// Double-buffered LDS + register prefetch: one barrier per K-step, global
// loads for step k+1 overlap the 8 WMMAs of step k.
//   energy[b,t] = sum_n w[n] * tanh(Whb[b,n] + Uv[b,t,n]); mask -> -inf
// ---------------------------------------------------------------------------
__global__ __launch_bounds__(256) void uv_energy_kernel(
    const float* __restrict__ keys, const float* __restrict__ U,
    const float* __restrict__ Whb,  const float* __restrict__ wvec,
    const unsigned char* __restrict__ masks, float* __restrict__ energies_out)
{
    __shared__ __bf16 As[2][MT  * LDA];   // 2 x  6 KB: keys tile (bf16)
    __shared__ __bf16 Bs[2][BN_ * LDA];   // 2 x 24 KB: U tile (bf16)
    __shared__ float  ener[MT];

    const int tid  = threadIdx.x;
    const int b    = blockIdx.x / (T_ / MT);
    const int t0   = (blockIdx.x % (T_ / MT)) * MT;

    if (tid < MT) ener[tid] = 0.f;

    const int wave = tid >> 5;           // 0..7
    const int lane = tid & 31;
    const int mi   = wave >> 1;          // M sub-tile 0..3 (16 rows each)
    const int nh   = wave & 1;           // N half: tiles 0..7 or 8..15

    f32x8 acc[8];
    #pragma unroll
    for (int j = 0; j < 8; ++j) acc[j] = f32x8{};

    const float* keyrow = keys + ((size_t)b * T_ + t0) * (size_t)KS_;

    // staging assignment: thread -> (row, 8-col group)
    const int srow = tid >> 2;                 // 0..63
    const int scg  = (tid & 3) * 8;            // 0,8,16,24
    const float* aSrc = keyrow + (size_t)srow * KS_ + scg;
    const float* bSrc[4];
    #pragma unroll
    for (int j = 0; j < 4; ++j)
        bSrc[j] = U + (size_t)(srow + 64 * j) * KS_ + scg;

    // Fragment addressing (contiguous k = 16h + s for A and B, not the ISA table order):
    // A (16x32 bf16): lane<16 -> K 0..15, lane>=16 -> K 16..31 (contiguous, matches B)
    const int am  = lane & 15;
    const int akb = (lane < 16) ? 0 : 16;
    // B (32x16 bf16): lane = N, lane<16 -> K 0..15, lane>=16 -> K 16..31 (contiguous)
    const int bnn = lane & 15;
    const int bkb = (lane < 16) ? 0 : 16;

    float4 pa0, pa1, pb0[4], pb1[4];

    // prologue: stage K-step 0 into buffer 0
    pa0 = *(const float4*)(aSrc);  pa1 = *(const float4*)(aSrc + 4);
    #pragma unroll
    for (int j = 0; j < 4; ++j) {
        pb0[j] = *(const float4*)(bSrc[j]);
        pb1[j] = *(const float4*)(bSrc[j] + 4);
    }
    cvt_store8(&As[0][srow * LDA + scg], pa0, pa1);
    #pragma unroll
    for (int j = 0; j < 4; ++j)
        cvt_store8(&Bs[0][(srow + 64 * j) * LDA + scg], pb0[j], pb1[j]);
    __syncthreads();

    for (int step = 0; step < NSTEP; ++step) {
        const int buf = step & 1;

        // issue global loads for next K-step early (overlap with WMMAs below)
        if (step + 1 < NSTEP) {
            const int kn = (step + 1) * KT;
            pa0 = *(const float4*)(aSrc + kn);  pa1 = *(const float4*)(aSrc + kn + 4);
            #pragma unroll
            for (int j = 0; j < 4; ++j) {
                pb0[j] = *(const float4*)(bSrc[j] + kn);
                pb1[j] = *(const float4*)(bSrc[j] + kn + 4);
            }
        }

        // fragments + 8 WMMAs from current buffer
        BF16Frag af;
        af.h[0] = *(const bf16x8*)&As[buf][(mi * 16 + am) * LDA + akb];
        af.h[1] = *(const bf16x8*)&As[buf][(mi * 16 + am) * LDA + akb + 8];
        #pragma unroll
        for (int j = 0; j < 8; ++j) {
            const int n0 = (nh * 8 + j) * 16 + bnn;
            BF16Frag bf;
            bf.h[0] = *(const bf16x8*)&Bs[buf][n0 * LDA + bkb];
            bf.h[1] = *(const bf16x8*)&Bs[buf][n0 * LDA + bkb + 8];
            acc[j] = __builtin_amdgcn_wmma_f32_16x16x32_bf16(
                false, af.v, false, bf.v, (short)0, acc[j], false, false);
        }

        // convert + store prefetched data into the other buffer
        if (step + 1 < NSTEP) {
            cvt_store8(&As[buf ^ 1][srow * LDA + scg], pa0, pa1);
            #pragma unroll
            for (int j = 0; j < 4; ++j)
                cvt_store8(&Bs[buf ^ 1][(srow + 64 * j) * LDA + scg], pb0[j], pb1[j]);
        }
        __syncthreads();
    }

    // Epilogue: C/D layout -> lane&15 = N col, VGPR r = M row (+8 for lanes>=16)
    const float* WhbB = Whb + (size_t)b * BN_;
    float part[8];
    #pragma unroll
    for (int r = 0; r < 8; ++r) part[r] = 0.f;
    #pragma unroll
    for (int j = 0; j < 8; ++j) {
        const int n  = (nh * 8 + j) * 16 + (lane & 15);
        const float wn = wvec[n];
        const float wh = WhbB[n];
        #pragma unroll
        for (int r = 0; r < 8; ++r)
            part[r] += wn * fast_tanh(wh + acc[j][r]);
    }
    #pragma unroll
    for (int r = 0; r < 8; ++r) {
        // reduce over the 16 lanes of each half (xor bits 0..3 keep halves separate)
        for (int off = 1; off < 16; off <<= 1)
            part[r] += __shfl_xor(part[r], off, 32);
        if ((lane & 15) == 0) {
            const int m = mi * 16 + ((lane >= 16) ? 8 : 0) + r;
            atomicAdd(&ener[m], part[r]);   // ds_add_f32; 2 waves per M slice
        }
    }
    __syncthreads();

    if (tid < MT) {
        const size_t gi = (size_t)b * T_ + t0 + tid;
        float e = ener[tid];
        if (masks[gi]) e = -INFINITY;
        energies_out[gi] = e;
    }
}

// ---------------------------------------------------------------------------
// Kernel 3: per-batch softmax over T, top-20 (tie-break low index), sparse
//           weighted gather of values -> attn_feats
// ---------------------------------------------------------------------------
__global__ __launch_bounds__(256) void softmax_topk_kernel(
    const float* __restrict__ energies, const float* __restrict__ values,
    float* __restrict__ weights_out, float* __restrict__ feats_out)
{
    __shared__ float se[T_];     // energies copy (destroyed by selection)
    __shared__ float sw[T_];     // softmax weights
    __shared__ float rv[256];
    __shared__ int   ri[256];
    __shared__ int   topidx[TOPK_];
    __shared__ float topw[TOPK_];

    const int b   = blockIdx.x;
    const int tid = threadIdx.x;
    const float* eb = energies + (size_t)b * T_;

    // max
    float lmax = -INFINITY;
    for (int t = tid; t < T_; t += 256) { float e = eb[t]; se[t] = e; lmax = fmaxf(lmax, e); }
    rv[tid] = lmax; __syncthreads();
    for (int s = 128; s > 0; s >>= 1) { if (tid < s) rv[tid] = fmaxf(rv[tid], rv[tid + s]); __syncthreads(); }
    const float gmax = rv[0]; __syncthreads();

    // sum exp
    float lsum = 0.f;
    for (int t = tid; t < T_; t += 256) { float p = fast_exp(se[t] - gmax); sw[t] = p; lsum += p; }
    rv[tid] = lsum; __syncthreads();
    for (int s = 128; s > 0; s >>= 1) { if (tid < s) rv[tid] += rv[tid + s]; __syncthreads(); }
    const float inv = 1.0f / rv[0]; __syncthreads();

    for (int t = tid; t < T_; t += 256) {
        const float wgt = sw[t] * inv;
        sw[t] = wgt;
        weights_out[(size_t)b * T_ + t] = wgt;
    }
    __syncthreads();

    // top-20 by energy (== top-20 by weight), lower index wins ties
    for (int it = 0; it < TOPK_; ++it) {
        float bv = -INFINITY; int bi = T_;
        for (int t = tid; t < T_; t += 256) {
            const float v = se[t];
            if (v > bv || (v == bv && t < bi)) { bv = v; bi = t; }
        }
        rv[tid] = bv; ri[tid] = bi; __syncthreads();
        for (int s = 128; s > 0; s >>= 1) {
            if (tid < s) {
                const float ov = rv[tid + s]; const int oi = ri[tid + s];
                if (ov > rv[tid] || (ov == rv[tid] && oi < ri[tid])) { rv[tid] = ov; ri[tid] = oi; }
            }
            __syncthreads();
        }
        if (tid == 0) { const int sel = ri[0]; topidx[it] = sel; topw[it] = sw[sel]; se[sel] = -INFINITY; }
        __syncthreads();
    }

    // attn_feats[b,d] = sum_k topw[k] * values[b, topidx[k], d]
    for (int d = tid; d < DV_; d += 256) {
        float a = 0.f;
        #pragma unroll
        for (int k = 0; k < TOPK_; ++k)
            a += topw[k] * values[((size_t)b * T_ + topidx[k]) * DV_ + d];
        feats_out[(size_t)b * DV_ + d] = a;
    }
}

// ---------------------------------------------------------------------------
extern "C" void kernel_launch(void* const* d_in, const int* in_sizes, int n_in,
                              void* d_out, int out_size, void* d_ws, size_t ws_size,
                              hipStream_t stream) {
    const float* query  = (const float*)d_in[0];
    const float* keys   = (const float*)d_in[1];
    const float* values = (const float*)d_in[2];
    const float* W      = (const float*)d_in[3];
    const float* U      = (const float*)d_in[4];
    const float* bias   = (const float*)d_in[5];
    const float* wvec   = (const float*)d_in[6];
    const unsigned char* masks = (const unsigned char*)d_in[7];

    // d_out = [attn_feats (B*DV) | weights (B*T) | energies (B*T)]
    float* feats    = (float*)d_out;
    float* weights  = feats   + (size_t)B_ * DV_;
    float* energies = weights + (size_t)B_ * T_;

    float* Whb = (float*)d_ws;   // B*BN fp32 = 32 KB scratch

    whb_kernel<<<B_, 256, 0, stream>>>(query, W, bias, Whb);
    uv_energy_kernel<<<B_ * (T_ / MT), 256, 0, stream>>>(keys, U, Whb, wvec, masks, energies);
    softmax_topk_kernel<<<B_, 256, 0, stream>>>(energies, values, weights, feats);
}
